// Encoder_33758442947336
// MI455X (gfx1250) — matrix operand order repaired and checked by reading, not compiled, not run on hardware
//
#include <hip/hip_runtime.h>
#include <hip/hip_fp16.h>

typedef _Float16 f16;
typedef _Float16 v4h  __attribute__((ext_vector_type(4)));
typedef _Float16 v8h  __attribute__((ext_vector_type(8)));
typedef _Float16 v16h __attribute__((ext_vector_type(16)));
typedef float    v8f  __attribute__((ext_vector_type(8)));

union V16 { v16h v; v8h h[2]; };

// ---------------- pack OIHW f32 weights into WMMA B-fragment tiles ----------------
// dst layout: ((ct*9 + tap)*KC + kc)*512 + n*32 + k   (n = co within tile, k = ci within chunk)
__global__ void k_pack_w(const float* __restrict__ src, f16* __restrict__ dst,
                         int Co, int Ci, int total) {
    int i = blockIdx.x * blockDim.x + threadIdx.x;
    if (i >= total) return;
    int KC  = (Ci + 31) >> 5;
    int k   = i & 31;
    int n   = (i >> 5) & 15;
    int t   = i >> 9;            // (ct*9 + tap)*KC + kc
    int kc  = t % KC;
    int tap = (t / KC) % 9;
    int ct  = t / (KC * 9);
    int ci  = kc * 32 + k;
    int co  = ct * 16 + n;
    f16 v = (f16)0;
    if (ci < Ci) v = (f16)src[((size_t)co * Ci + ci) * 9 + tap];
    dst[i] = v;
}

// ---------------- conv0: 7x7 reflect-pad, NCHW f32 in -> NHWC f32 out (3 -> 16) ----------------
__global__ void k_conv0(const float* __restrict__ x, const float* __restrict__ w,
                        float* __restrict__ y, int B, int H, int W) {
    long long idx = (long long)blockIdx.x * blockDim.x + threadIdx.x;  // (b,oy,ox)
    long long tot = (long long)B * H * W;
    if (idx >= tot) return;
    int ox = (int)(idx % W);
    int oy = (int)((idx / W) % H);
    int b  = (int)(idx / ((long long)W * H));
    float acc[16];
#pragma unroll
    for (int co = 0; co < 16; co++) acc[co] = 0.f;
    for (int ky = 0; ky < 7; ky++) {
        int iy = oy - 3 + ky; iy = iy < 0 ? -iy : (iy >= H ? 2 * H - 2 - iy : iy);
        for (int kx = 0; kx < 7; kx++) {
            int ix = ox - 3 + kx; ix = ix < 0 ? -ix : (ix >= W ? 2 * W - 2 - ix : ix);
#pragma unroll
            for (int c = 0; c < 3; c++) {
                float v = x[((long long)(b * 3 + c) * H + iy) * W + ix];
#pragma unroll
                for (int co = 0; co < 16; co++)
                    acc[co] += v * w[((co * 3 + c) * 7 + ky) * 7 + kx];
            }
        }
    }
    float* yp = y + idx * 16;
#pragma unroll
    for (int co = 0; co < 16; co++) yp[co] = acc[co];
}

// ---------------- instance-norm stats, coalesced: block = (b, group of 16 channels) ----------------
__global__ void k_stats16(const float* __restrict__ y, float* __restrict__ mean,
                          float* __restrict__ rstd, int HW, int C) {
    int CG = C >> 4;
    int cg = blockIdx.x % CG;
    int b  = blockIdx.x / CG;
    int cl = threadIdx.x & 15;       // channel within group
    int pr = threadIdx.x >> 4;       // pixel row 0..15
    const float* p = y + (size_t)b * HW * C + cg * 16 + cl;
    float s = 0.f, s2 = 0.f;
    for (int i = pr; i < HW; i += 16) {
        float v = p[(size_t)i * C];
        s += v; s2 += v * v;
    }
    __shared__ float sh[256], sh2[256];
    sh[cl * 16 + pr] = s; sh2[cl * 16 + pr] = s2;
    __syncthreads();
    for (int o = 8; o > 0; o >>= 1) {
        if (pr < o) {
            sh[cl * 16 + pr]  += sh[cl * 16 + pr + o];
            sh2[cl * 16 + pr] += sh2[cl * 16 + pr + o];
        }
        __syncthreads();
    }
    if (pr == 0) {
        float m = sh[cl * 16] / (float)HW;
        float v = sh2[cl * 16] / (float)HW - m * m;
        int bc = b * C + cg * 16 + cl;
        mean[bc] = m;
        rstd[bc] = rsqrtf(fmaxf(v, 0.f) + 1e-5f);
    }
}

// ---------------- zero 1-pixel halo border of (B, H+2, W+2, C) f16 buffer ----------------
__global__ void k_zero_halo(f16* __restrict__ buf, int B, int H, int W, int C) {
    int P = 2 * (W + 2) + 2 * H;             // border pixels per image
    int CV = C >> 3;                          // 8-wide channel vectors
    long long tot = (long long)B * P * CV;
    long long i = (long long)blockIdx.x * blockDim.x + threadIdx.x;
    if (i >= tot) return;
    int cv = (int)(i % CV);
    long long t = i / CV;
    int j = (int)(t % P);
    int b = (int)(t / P);
    int y, xq;
    if (j < W + 2)            { y = 0;      xq = j; }
    else if (j < 2 * (W + 2)) { y = H + 1;  xq = j - (W + 2); }
    else {
        int r = j - 2 * (W + 2);
        y  = 1 + (r >> 1);
        xq = (r & 1) ? (W + 1) : 0;
    }
    v8h z = {};
    *(v8h*)(buf + (((long long)b * (H + 2) + y) * (W + 2) + xq) * C + cv * 8) = z;
}

// ---------------- normalize + ReLU + cast f16 into haloed NHWC, 4 elements/thread ----------------
__global__ void k_norm_relu(const float* __restrict__ y, const float* __restrict__ mean,
                            const float* __restrict__ rstd, f16* __restrict__ out,
                            long long n4, int C, int H, int W) {
    long long i = (long long)blockIdx.x * blockDim.x + threadIdx.x;
    if (i >= n4) return;
    long long i4 = i * 4;
    int c0 = (int)(i4 % C);
    long long pix = i4 / C;
    int xx = (int)(pix % W);
    int yy = (int)((pix / W) % H);
    int b  = (int)(pix / ((long long)W * H));
    const float* mp = mean + b * C + c0;
    const float* rp = rstd + b * C + c0;
    float4 v = *(const float4*)(y + i4);
    v4h o;
    o.x = (f16)fmaxf((v.x - mp[0]) * rp[0], 0.f);
    o.y = (f16)fmaxf((v.y - mp[1]) * rp[1], 0.f);
    o.z = (f16)fmaxf((v.z - mp[2]) * rp[2], 0.f);
    o.w = (f16)fmaxf((v.w - mp[3]) * rp[3], 0.f);
    *(v4h*)(out + (((long long)b * (H + 2) + yy + 1) * (W + 2) + xx + 1) * C + c0) = o;
}

// ---------------- 3x3 stride-2 pad-1 conv, haloed NHWC f16 -> NHWC f32, WMMA ----------------
template <int KC, bool CI16, bool TWO>
__global__ void k_conv3x3s2_wmma(const f16* __restrict__ x, const f16* __restrict__ wp,
                                 float* __restrict__ y,
                                 int B, int Hi, int Wi, int Ho, int Wo, int Co) {
    constexpr int CI = CI16 ? 16 : KC * 32;
    const int lane = threadIdx.x & 31;
    const int wave = threadIdx.x >> 5;
    const int ct0  = blockIdx.y * 2;
    const long long mt  = (long long)blockIdx.x * 4 + wave;
    const long long tot = (long long)B * Ho * Wo;
    const int m = lane & 15;
    const long long p0 = mt * 16 + m;
    const long long p  = p0 < tot ? p0 : 0;       // clamp (halo makes loads safe)
    const int ox = (int)(p % Wo);
    const int oy = (int)((p / Wo) % Ho);
    const int b  = (int)(p / ((long long)Wo * Ho));
    const int kbase = (lane >> 4) * 8;
    const size_t ctStride = (size_t)9 * KC * 512;
    const f16* wb = wp + (size_t)ct0 * ctStride + (size_t)(lane & 15) * 32 + (size_t)(lane >> 4) * 8;
    const v8h zv = {};
    v8f d0 = {}, d1 = {};
#pragma unroll
    for (int ky = 0; ky < 3; ky++) {
        const int iy = oy * 2 + ky;               // haloed row index
#pragma unroll
        for (int kx = 0; kx < 3; kx++) {
            const int ix = ox * 2 + kx;           // haloed col index
            const f16* xp = x + ((long long)(b * (Hi + 2) + iy) * (Wi + 2) + ix) * CI;
            const f16* wt = wb + (size_t)(ky * 3 + kx) * KC * 512;
#pragma unroll
            for (int kc = 0; kc < KC; kc++) {
                V16 a;
                a.h[0] = *(const v8h*)(xp + kc * 32 + kbase);
                if (CI16) a.h[1] = zv;
                else      a.h[1] = *(const v8h*)(xp + kc * 32 + kbase + 16);
                if (kc + 1 < KC)
                    __builtin_prefetch((const void*)(xp + kc * 32 + kbase + 32), 0, 1);
                V16 bb0; bb0.h[0] = *(const v8h*)(wt + (size_t)kc * 512); bb0.h[1] = *(const v8h*)(wt + (size_t)kc * 512 + 16); v16h b0 = bb0.v;
                d0 = __builtin_amdgcn_wmma_f32_16x16x32_f16(false, a.v, false, b0,
                                                            (short)0, d0, false, false);
                if (TWO) {
                    V16 bb1; bb1.h[0] = *(const v8h*)(wt + ctStride + (size_t)kc * 512); bb1.h[1] = *(const v8h*)(wt + ctStride + (size_t)kc * 512 + 16); v16h b1 = bb1.v;
                    d1 = __builtin_amdgcn_wmma_f32_16x16x32_f16(false, a.v, false, b1,
                                                                (short)0, d1, false, false);
                }
            }
        }
    }
#pragma unroll
    for (int r = 0; r < 8; r++) {
        int ms = r + (lane >> 4) * 8;
        long long ps = mt * 16 + ms;
        if (ps < tot) {
            float* yp = y + ps * Co + ct0 * 16 + (lane & 15);
            yp[0] = d0[r];
            if (TWO) yp[16] = d1[r];
        }
    }
}

// ---------------- convT 3x3 s2 p1 op1 (gather, per-parity), haloed in, WMMA ----------------
template <int KC, bool TWO>
__global__ void k_convT_wmma(const f16* __restrict__ x, const f16* __restrict__ wp,
                             float* __restrict__ y,
                             int B, int Hi, int Wi, int Co) {
    constexpr int CI = KC * 32;
    const int lane = threadIdx.x & 31;
    const int wave = threadIdx.x >> 5;
    const int ct0  = blockIdx.y * 2;
    const int py = (int)blockIdx.z >> 1, px = (int)blockIdx.z & 1;
    const int Ho = 2 * Hi, Wo = 2 * Wi;
    const long long mt  = (long long)blockIdx.x * 4 + wave;
    const long long tot = (long long)B * Hi * Wi;   // pixels per parity class
    const int m = lane & 15;
    const long long q0 = mt * 16 + m;
    const long long q  = q0 < tot ? q0 : 0;         // clamp
    const int cx = (int)(q % Wi);
    const int cy = (int)((q / Wi) % Hi);
    const int b  = (int)(q / ((long long)Wi * Hi));
    const int oy = 2 * cy + py, ox = 2 * cx + px;
    int kys[2], kxs[2], nky, nkx;
    if (py == 0) { kys[0] = 1; kys[1] = 1; nky = 1; } else { kys[0] = 0; kys[1] = 2; nky = 2; }
    if (px == 0) { kxs[0] = 1; kxs[1] = 1; nkx = 1; } else { kxs[0] = 0; kxs[1] = 2; nkx = 2; }
    const int kbase = (lane >> 4) * 8;
    const size_t ctStride = (size_t)9 * KC * 512;
    const f16* wb = wp + (size_t)ct0 * ctStride + (size_t)(lane & 15) * 32 + (size_t)(lane >> 4) * 8;
    v8f d0 = {}, d1 = {};
    for (int ti = 0; ti < nky; ti++) {
        const int ky = kys[ti];
        const int iy = ((oy - 1 + ky) >> 1) + 1;    // haloed row; in [1, Hi+1]
        for (int tj = 0; tj < nkx; tj++) {
            const int kx = kxs[tj];
            const int ix = ((ox - 1 + kx) >> 1) + 1;
            const f16* xp = x + ((long long)(b * (Hi + 2) + iy) * (Wi + 2) + ix) * CI;
            const f16* wt = wb + (size_t)(ky * 3 + kx) * KC * 512;
#pragma unroll
            for (int kc = 0; kc < KC; kc++) {
                V16 a;
                a.h[0] = *(const v8h*)(xp + kc * 32 + kbase);
                a.h[1] = *(const v8h*)(xp + kc * 32 + kbase + 16);
                if (kc + 1 < KC)
                    __builtin_prefetch((const void*)(xp + kc * 32 + kbase + 32), 0, 1);
                V16 bb0; bb0.h[0] = *(const v8h*)(wt + (size_t)kc * 512); bb0.h[1] = *(const v8h*)(wt + (size_t)kc * 512 + 16); v16h b0 = bb0.v;
                d0 = __builtin_amdgcn_wmma_f32_16x16x32_f16(false, a.v, false, b0,
                                                            (short)0, d0, false, false);
                if (TWO) {
                    V16 bb1; bb1.h[0] = *(const v8h*)(wt + ctStride + (size_t)kc * 512); bb1.h[1] = *(const v8h*)(wt + ctStride + (size_t)kc * 512 + 16); v16h b1 = bb1.v;
                    d1 = __builtin_amdgcn_wmma_f32_16x16x32_f16(false, a.v, false, b1,
                                                                (short)0, d1, false, false);
                }
            }
        }
    }
#pragma unroll
    for (int r = 0; r < 8; r++) {
        int ms = r + (lane >> 4) * 8;
        long long qs = mt * 16 + ms;
        if (qs < tot) {
            int cxs = (int)(qs % Wi);
            int cys = (int)((qs / Wi) % Hi);
            int bs  = (int)(qs / ((long long)Wi * Hi));
            long long oys = 2 * cys + py, oxs = 2 * cxs + px;
            float* yp = y + (((long long)bs * Ho + oys) * Wo + oxs) * Co + ct0 * 16 + (lane & 15);
            yp[0] = d0[r];
            if (TWO) yp[16] = d1[r];
        }
    }
}

// ---------------- final conv 7x7 (16->3, haloed input) + tanh + instance accumulation ----------------
__global__ void k_final(const f16* __restrict__ x, const float* __restrict__ w,
                        const float* __restrict__ bias, const int* __restrict__ inst,
                        float* __restrict__ sums, float* __restrict__ cnt,
                        int B, int H, int W) {
    long long idx = (long long)blockIdx.x * blockDim.x + threadIdx.x;
    long long tot = (long long)B * H * W;
    if (idx >= tot) return;
    int ox = (int)(idx % W);
    int oy = (int)((idx / W) % H);
    int b  = (int)(idx / ((long long)W * H));
    float acc[3] = { bias[0], bias[1], bias[2] };
    for (int ky = 0; ky < 7; ky++) {
        int iy = oy - 3 + ky; iy = iy < 0 ? -iy : (iy >= H ? 2 * H - 2 - iy : iy);
        for (int kx = 0; kx < 7; kx++) {
            int ix = ox - 3 + kx; ix = ix < 0 ? -ix : (ix >= W ? 2 * W - 2 - ix : ix);
            const f16* xp = x + (((long long)b * (H + 2) + iy + 1) * (W + 2) + ix + 1) * 16;
            V16 xv;
            xv.h[0] = *(const v8h*)xp;
            xv.h[1] = *(const v8h*)(xp + 8);
#pragma unroll
            for (int c = 0; c < 16; c++) {
                float v = (float)xv.v[c];
#pragma unroll
                for (int co = 0; co < 3; co++)
                    acc[co] += v * w[((co * 16 + c) * 7 + ky) * 7 + kx];
            }
        }
    }
    int id = inst[(long long)b * H * W + (long long)oy * W + ox];
    float* sp = sums + (long long)(b * 32 + id) * 3;
#pragma unroll
    for (int co = 0; co < 3; co++) atomicAdd(&sp[co], tanhf(acc[co]));
    atomicAdd(&cnt[b * 32 + id], 1.0f);
}

__global__ void k_pool_mean(float* sums, const float* cnt, int n) {  // n = B*32
    int i = blockIdx.x * blockDim.x + threadIdx.x;
    if (i >= n) return;
    float c = fmaxf(cnt[i], 1.0f);
    sums[i * 3 + 0] /= c; sums[i * 3 + 1] /= c; sums[i * 3 + 2] /= c;
}

__global__ void k_bcast(const float* __restrict__ means, const int* __restrict__ inst,
                        float* __restrict__ out, int B, int H, int W) {
    long long idx = (long long)blockIdx.x * blockDim.x + threadIdx.x;
    long long tot = (long long)B * 3 * H * W;
    if (idx >= tot) return;
    long long hw = idx % ((long long)H * W);
    int c = (int)((idx / ((long long)H * W)) % 3);
    int b = (int)(idx / ((long long)3 * H * W));
    int id = inst[(long long)b * H * W + hw];
    out[idx] = means[(long long)(b * 32 + id) * 3 + c];
}

// ---------------- host orchestration ----------------
static inline void stats_norm(const float* A32, float* mean, float* rstd, f16* F16a,
                              int B, int H, int C, hipStream_t stream) {
    k_stats16<<<B * (C / 16), 256, 0, stream>>>(A32, mean, rstd, H * H, C);
    long long halo = (long long)B * (2 * (H + 2) + 2 * H) * (C / 8);
    k_zero_halo<<<dim3((unsigned)((halo + 255) / 256)), 256, 0, stream>>>(F16a, B, H, H, C);
    long long n4 = (long long)B * H * H * C / 4;
    k_norm_relu<<<dim3((unsigned)((n4 + 255) / 256)), 256, 0, stream>>>(A32, mean, rstd, F16a, n4, C, H, H);
}

extern "C" void kernel_launch(void* const* d_in, const int* in_sizes, int n_in,
                              void* d_out, int out_size, void* d_ws, size_t ws_size,
                              hipStream_t stream) {
    (void)in_sizes; (void)n_in; (void)out_size; (void)ws_size;
    const int B = 4, H0 = 512;

    const float* x    = (const float*)d_in[0];
    const int*   inst = (const int*)d_in[1];
    const float* w0   = (const float*)d_in[2];   // (16,3,7,7); hidden-layer biases absorbed by InstanceNorm
    const float* dwf[4] = { (const float*)d_in[4],  (const float*)d_in[6],
                            (const float*)d_in[8],  (const float*)d_in[10] };
    const float* uwf[4] = { (const float*)d_in[12], (const float*)d_in[14],
                            (const float*)d_in[16], (const float*)d_in[18] };
    const float* wfin = (const float*)d_in[20];  // (3,16,7,7)
    const float* bfin = (const float*)d_in[21];

    char* ws = (char*)d_ws;
    float* A32  = (float*)ws;                                  // 67.1 MB f32 conv outputs
    f16*   F16a = (f16*)(ws + (size_t)67108864);               // 33.9 MB haloed f16 acts
    f16*   W16  = (f16*)(ws + (size_t)101187584);              // packed f16 weights (~1.6 MB)
    float* mean = (float*)(ws + (size_t)103284736);
    float* rstd = mean + 2048;
    float* sums = rstd + 2048;                                 // 384 floats, then 128 counts
    float* cnt  = sums + 384;

    // ---- pack mid-layer weights into WMMA fragment tiles ----
    f16* dWp[4]; f16* uWp[4];
    size_t off = 0;
    for (int i = 0; i < 4; i++) {
        int Ci = 16 << i, Co = 32 << i, KC = (Ci + 31) >> 5;
        int n = (Co / 16) * 9 * KC * 512;
        dWp[i] = W16 + off; off += (size_t)n;
        k_pack_w<<<(n + 255) / 256, 256, 0, stream>>>(dwf[i], dWp[i], Co, Ci, n);
    }
    for (int i = 0; i < 4; i++) {
        int Ci = 256 >> i, Co = 128 >> i, KC = Ci >> 5;
        int n = (Co / 16) * 9 * KC * 512;
        uWp[i] = W16 + off; off += (size_t)n;
        k_pack_w<<<(n + 255) / 256, 256, 0, stream>>>(uwf[i], uWp[i], Co, Ci, n);
    }

    // ---- conv0 + IN + ReLU ----
    long long tot0 = (long long)B * H0 * H0;
    k_conv0<<<dim3((unsigned)((tot0 + 255) / 256)), 256, 0, stream>>>(x, w0, A32, B, H0, H0);
    stats_norm(A32, mean, rstd, F16a, B, H0, 16, stream);

    // ---- downsampling stack (WMMA) ----
    // layer i: (H, C) -> (H/2, 2C);  H = 512>>i, C = 16<<i
    {
        auto gx = [&](int Ho) { return (unsigned)(((long long)B * Ho * Ho / 16 + 3) / 4); };
        k_conv3x3s2_wmma<1, true,  true><<<dim3(gx(256), 1), 128, 0, stream>>>(F16a, dWp[0], A32, B, 512, 512, 256, 256, 32);
        stats_norm(A32, mean, rstd, F16a, B, 256, 32, stream);
        k_conv3x3s2_wmma<1, false, true><<<dim3(gx(128), 2), 128, 0, stream>>>(F16a, dWp[1], A32, B, 256, 256, 128, 128, 64);
        stats_norm(A32, mean, rstd, F16a, B, 128, 64, stream);
        k_conv3x3s2_wmma<2, false, true><<<dim3(gx(64), 4), 128, 0, stream>>>(F16a, dWp[2], A32, B, 128, 128, 64, 64, 128);
        stats_norm(A32, mean, rstd, F16a, B, 64, 128, stream);
        k_conv3x3s2_wmma<4, false, true><<<dim3(gx(32), 8), 128, 0, stream>>>(F16a, dWp[3], A32, B, 64, 64, 32, 32, 256);
        stats_norm(A32, mean, rstd, F16a, B, 32, 256, stream);
    }

    // ---- upsampling stack (WMMA, gather convT per parity class) ----
    // layer i: (H, C) -> (2H, C/2);  H = 32<<i, C = 256>>i
    {
        auto gx = [&](int Hi) { return (unsigned)(((long long)B * Hi * Hi / 16 + 3) / 4); };
        k_convT_wmma<8, true ><<<dim3(gx(32), 4, 4), 128, 0, stream>>>(F16a, uWp[0], A32, B, 32, 32, 128);
        stats_norm(A32, mean, rstd, F16a, B, 64, 128, stream);
        k_convT_wmma<4, true ><<<dim3(gx(64), 2, 4), 128, 0, stream>>>(F16a, uWp[1], A32, B, 64, 64, 64);
        stats_norm(A32, mean, rstd, F16a, B, 128, 64, stream);
        k_convT_wmma<2, true ><<<dim3(gx(128), 1, 4), 128, 0, stream>>>(F16a, uWp[2], A32, B, 128, 128, 32);
        stats_norm(A32, mean, rstd, F16a, B, 256, 32, stream);
        k_convT_wmma<1, false><<<dim3(gx(256), 1, 4), 128, 0, stream>>>(F16a, uWp[3], A32, B, 256, 256, 16);
        stats_norm(A32, mean, rstd, F16a, B, 512, 16, stream);
    }

    // ---- final conv + tanh fused with instance pooling ----
    hipMemsetAsync(sums, 0, (384 + 128) * sizeof(float), stream);
    k_final<<<dim3((unsigned)((tot0 + 255) / 256)), 256, 0, stream>>>(F16a, wfin, bfin, inst, sums, cnt, B, H0, H0);
    k_pool_mean<<<1, 128, 0, stream>>>(sums, cnt, B * 32);
    long long no = tot0 * 3;
    k_bcast<<<dim3((unsigned)((no + 255) / 256)), 256, 0, stream>>>(sums, inst, (float*)d_out, B, H0, H0);
}
